// HybridBlock_15702400434381
// MI455X (gfx1250) — matrix operand order repaired and checked by reading, not compiled, not run on hardware
//
#include <hip/hip_runtime.h>

// ---------------------------------------------------------------------------
// Types for CDNA5 WMMA (wave32): v_wmma_f32_16x16x32_bf16
// ---------------------------------------------------------------------------
typedef __attribute__((ext_vector_type(16))) __bf16 v16bf;
typedef __attribute__((ext_vector_type(8)))  __bf16 v8bf;
typedef __attribute__((ext_vector_type(4)))  __bf16 v4bf;
typedef __attribute__((ext_vector_type(8)))  float  v8f;
typedef __attribute__((ext_vector_type(4)))  int    v4i;

#define D_   1024
#define S_   2048
#define H_   16
#define DH_  64
#define WIN_ 128
#define NTOK 4096   // B*S

#if defined(__HIP_DEVICE_COMPILE__) && __has_builtin(__builtin_amdgcn_global_load_async_to_lds_b128)
#define HAVE_ASYNC_LDS 1
#else
#define HAVE_ASYNC_LDS 0
#endif

#if HAVE_ASYNC_LDS
typedef __attribute__((address_space(1))) v4i GASv4i;   // global int4
typedef __attribute__((address_space(3))) v4i LASv4i;   // LDS int4
#endif

// 16B global -> LDS copy: async DMA path on gfx1250, sync fallback otherwise.
__device__ __forceinline__ void copy16_g2l(const __bf16* g, __bf16* l) {
#if HAVE_ASYNC_LDS
  __builtin_amdgcn_global_load_async_to_lds_b128((GASv4i*)g, (LASv4i*)l, 0, 0);
#else
  *(v8bf*)l = *(const v8bf*)g;
#endif
}
__device__ __forceinline__ void async_wait_le4() {
#if HAVE_ASYNC_LDS
  asm volatile("s_wait_asynccnt 0x4" ::: "memory");
#endif
}
__device__ __forceinline__ void async_wait_0() {
#if HAVE_ASYNC_LDS
  asm volatile("s_wait_asynccnt 0x0" ::: "memory");
#endif
}

__device__ __forceinline__ v8f wmma_bf16(v16bf a, v16bf b, v8f c) {
  return __builtin_amdgcn_wmma_f32_16x16x32_bf16(false, a, false, b, (short)0, c,
                                                 false, false);
}

// A-matrix 16x32 bf16 fragment (ISA 7.12.2): lane L holds row M=L%16,
// K runs [8h..8h+7] and [16+8h..16+8h+7] where h=L/16.  rowp -> 32-wide K row.
__device__ __forceinline__ v16bf make_afrag(const __bf16* rowp, int half) {
  v8bf lo = *(const v8bf*)(rowp + half * 8);
  v8bf hi = *(const v8bf*)(rowp + 16 + half * 8);
  v16bf a;
#pragma unroll
  for (int i = 0; i < 8; ++i) { a[i] = lo[i]; a[i + 8] = hi[i]; }
  return a;
}

// B-matrix 32x16 bf16 fragment: lane L holds column N=L%16,
// K runs [8h..8h+7] and [16+8h..16+8h+7] where h=L/16.  colp -> 32-wide K row for column N.
__device__ __forceinline__ v16bf make_bfrag(const __bf16* colp, int half) {
  v8bf lo = *(const v8bf*)(colp + half * 8);
  v8bf hi = *(const v8bf*)(colp + 16 + half * 8);
  v16bf b;
#pragma unroll
  for (int i = 0; i < 8; ++i) { b[i] = lo[i]; b[i + 8] = hi[i]; }
  return b;
}

// Stage one 128x32 A panel + 128x32 B panel into LDS (4 x b128 per thread).
__device__ __forceinline__ void gemm_stage(const __bf16* __restrict__ A,
                                           const __bf16* __restrict__ B,
                                           __bf16 (*Asm)[40], __bf16 (*Bsm)[40],
                                           int m0, int n0, int k0, int t) {
#pragma unroll
  for (int i = 0; i < 2; ++i) {
    int id = t + 256 * i;
    int r = id >> 2, c8 = (id & 3) * 8;
    copy16_g2l(&A[(size_t)(m0 + r) * 1024 + k0 + c8], &Asm[r][c8]);
    copy16_g2l(&B[(size_t)(n0 + r) * 1024 + k0 + c8], &Bsm[r][c8]);
  }
}

// ---------------------------------------------------------------------------
// fp32 -> bf16 bulk convert
// ---------------------------------------------------------------------------
__global__ __launch_bounds__(256) void cvt_bf16(const float* __restrict__ in,
                                                __bf16* __restrict__ out, int n) {
  int i = (blockIdx.x * 256 + threadIdx.x) * 4;
  if (i + 3 < n) {
    float4 v = *(const float4*)(in + i);
    v4bf o; o[0] = (__bf16)v.x; o[1] = (__bf16)v.y; o[2] = (__bf16)v.z; o[3] = (__bf16)v.w;
    *(v4bf*)(out + i) = o;
  }
}

// ---------------------------------------------------------------------------
// QKV GEMM: C[4096,3072] = xb[4096,1024] * w1b[3072,1024]^T + bias
// Double-buffered async LDS staging; scatters into Q/K/V [B,H,S,DH] bf16;
// Q pre-scaled by DH^-0.5.  WG tile 128x128, 8 waves (4x2), K panel 32.
// ---------------------------------------------------------------------------
__global__ __launch_bounds__(256) void qkv_gemm(const __bf16* __restrict__ xb,
                                                const __bf16* __restrict__ w1b,
                                                const float* __restrict__ bias,
                                                __bf16* __restrict__ Qb,
                                                __bf16* __restrict__ Kb,
                                                __bf16* __restrict__ Vb) {
  const int m0 = (blockIdx.x / 24) * 128;
  const int n0 = (blockIdx.x % 24) * 128;
  __shared__ __align__(16) __bf16 Asm[2][128][40];
  __shared__ __align__(16) __bf16 Bsm[2][128][40];
  const int t = threadIdx.x;
  const int w = t >> 5, L = t & 31;
  const int wm = w >> 1, wn = w & 1;
  const int half = L >> 4, lr = L & 15;

  v8f acc[2][4] = {};
  gemm_stage(xb, w1b, Asm[0], Bsm[0], m0, n0, 0, t);          // prologue
  for (int k0 = 0; k0 < 1024; k0 += 32) {
    const int buf = (k0 >> 5) & 1;
    if (k0) __syncthreads();                                  // readers of buf^1 done
    if (k0 + 32 < 1024) {
      gemm_stage(xb, w1b, Asm[buf ^ 1], Bsm[buf ^ 1], m0, n0, k0 + 32, t);
      async_wait_le4();                                       // stage k resident
    } else {
      async_wait_0();
    }
    __syncthreads();
    v16bf afr[2], bfr[4];
#pragma unroll
    for (int i = 0; i < 2; ++i) afr[i] = make_afrag(&Asm[buf][wm * 32 + i * 16 + lr][0], half);
#pragma unroll
    for (int j = 0; j < 4; ++j) bfr[j] = make_bfrag(&Bsm[buf][wn * 64 + j * 16 + lr][0], half);
#pragma unroll
    for (int i = 0; i < 2; ++i)
#pragma unroll
      for (int j = 0; j < 4; ++j) acc[i][j] = wmma_bf16(afr[i], bfr[j], acc[i][j]);
  }
  // epilogue: C-layout lane holds col N=lr, rows 8*half+r
#pragma unroll
  for (int j = 0; j < 4; ++j) {
    const int C = n0 + wn * 64 + j * 16 + lr;
    const int which = C >> 10;
    const int dd = C & 1023;
    const int h = dd >> 6, dh = dd & 63;
    const float bs = bias[C];
#pragma unroll
    for (int i = 0; i < 2; ++i)
#pragma unroll
      for (int r = 0; r < 8; ++r) {
        const int R = m0 + wm * 32 + i * 16 + half * 8 + r;
        const int b = R >> 11, s = R & 2047;
        const size_t idx = (((size_t)(b * H_ + h)) * S_ + s) * DH_ + dh;
        const float val = acc[i][j][r] + bs;
        if (which == 0)      Qb[idx] = (__bf16)(val * 0.125f);   // * DH^-0.5
        else if (which == 1) Kb[idx] = (__bf16)val;
        else                 Vb[idx] = (__bf16)val;
      }
  }
}

// ---------------------------------------------------------------------------
// Banded flash attention.  One WG per (b,h,128-query block), 8 waves,
// each wave owns 16 query rows.  Keys streamed in 32-wide LDS chunks over
// [q0-128, q0+255].  Online softmax; P via per-wave LDS C->A relayout.
// ---------------------------------------------------------------------------
__global__ __launch_bounds__(256) void attn_kernel(const __bf16* __restrict__ Qb,
                                                   const __bf16* __restrict__ Kb,
                                                   const __bf16* __restrict__ Vb,
                                                   __bf16* __restrict__ attn) {
  const int id = blockIdx.x;
  const int qblk = id & 15;
  const int h = (id >> 4) & 15;
  const int b = id >> 8;
  const int q0 = qblk * 128;
  const size_t bh = ((size_t)b * H_ + h) * S_;

  __shared__ __align__(16) __bf16 Qs[128][72];
  __shared__ __align__(16) __bf16 Ks[32][72];
  __shared__ __align__(16) __bf16 Vt[64][40];   // transposed V chunk [dh][key]
  __shared__ __align__(16) __bf16 Ps[8][16][40];

  const int t = threadIdx.x;
  const int w = t >> 5, L = t & 31;
  const int half = L >> 4, lr = L & 15;

#pragma unroll
  for (int i = 0; i < 4; ++i) {                 // load 128x64 Q tile
    int idq = t + 256 * i;
    int r = idq >> 3, c8 = (idq & 7) * 8;
    *(v8bf*)&Qs[r][c8] = *(const v8bf*)&Qb[(bh + q0 + r) * DH_ + c8];
  }
  __syncthreads();

  v16bf qf[2];
  qf[0] = make_afrag(&Qs[w * 16 + lr][0], half);    // dh 0..31
  qf[1] = make_afrag(&Qs[w * 16 + lr][32], half);   // dh 32..63

  v8f acc[4] = {};
  float rmax[8], rsum[8];
#pragma unroll
  for (int r = 0; r < 8; ++r) { rmax[r] = -3.0e38f; rsum[r] = 0.0f; }

  int ks_begin = q0 - WIN_;       if (ks_begin < 0) ks_begin = 0;
  int ks_end   = q0 + 128 + WIN_; if (ks_end > S_) ks_end = S_;
  const int qrow_base = q0 + w * 16 + half * 8;

  for (int ks = ks_begin; ks < ks_end; ks += 32) {
    __syncthreads();
    {                                            // stage K and V^T chunks
      int r = t >> 3, c8 = (t & 7) * 8;
      *(v8bf*)&Ks[r][c8] = *(const v8bf*)&Kb[(bh + ks + r) * DH_ + c8];
      v8bf vv = *(const v8bf*)&Vb[(bh + ks + r) * DH_ + c8];
#pragma unroll
      for (int e = 0; e < 8; ++e) Vt[c8 + e][r] = vv[e];
    }
    __syncthreads();

    // scores S = Q K^T for 16x32 key slab (two 16x16 C tiles, K-dim = dh)
    v8f sc[2] = {};
#pragma unroll
    for (int nt = 0; nt < 2; ++nt) {
      sc[nt] = wmma_bf16(qf[0], make_bfrag(&Ks[nt * 16 + lr][0], half), sc[nt]);
      sc[nt] = wmma_bf16(qf[1], make_bfrag(&Ks[nt * 16 + lr][32], half), sc[nt]);
    }

    // band mask + chunk row max
    float cmax[8];
#pragma unroll
    for (int r = 0; r < 8; ++r) {
      const int qg = qrow_base + r;
      float m_ = -3.0e38f;
#pragma unroll
      for (int nt = 0; nt < 2; ++nt) {
        const int kg = ks + nt * 16 + lr;
        const bool valid = (kg - qg <= WIN_) && (qg - kg <= WIN_);
        const float s_ = valid ? sc[nt][r] : -3.0e38f;
        sc[nt][r] = s_;
        m_ = fmaxf(m_, s_);
      }
#pragma unroll
      for (int m = 1; m < 16; m <<= 1) m_ = fmaxf(m_, __shfl_xor(m_, m, 32));
      cmax[r] = m_;
    }

    // online softmax update
    float p0[8], p1[8];
#pragma unroll
    for (int r = 0; r < 8; ++r) {
      const float mnew = fmaxf(rmax[r], cmax[r]);
      const float scal = __expf(rmax[r] - mnew);
      rmax[r] = mnew;
      const float a0 = (sc[0][r] <= -1.0e37f) ? 0.0f : __expf(sc[0][r] - mnew);
      const float a1 = (sc[1][r] <= -1.0e37f) ? 0.0f : __expf(sc[1][r] - mnew);
      p0[r] = a0; p1[r] = a1;
      float cs = a0 + a1;
#pragma unroll
      for (int m = 1; m < 16; m <<= 1) cs += __shfl_xor(cs, m, 32);
      rsum[r] = rsum[r] * scal + cs;
#pragma unroll
      for (int dt = 0; dt < 4; ++dt) acc[dt][r] *= scal;
    }

    // C-layout -> A-layout relayout for P through per-wave LDS slab
#pragma unroll
    for (int r = 0; r < 8; ++r) {
      Ps[w][half * 8 + r][lr]      = (__bf16)p0[r];
      Ps[w][half * 8 + r][16 + lr] = (__bf16)p1[r];
    }
    asm volatile("s_wait_dscnt 0x0" ::: "memory");   // wave-internal LDS RAW
    v16bf pf = make_afrag(&Ps[w][lr][0], half);

    // O += P V   (four 16x16 dh tiles, K-dim = 32 keys)
#pragma unroll
    for (int dt = 0; dt < 4; ++dt)
      acc[dt] = wmma_bf16(pf, make_bfrag(&Vt[dt * 16 + lr][0], half), acc[dt]);
  }

  // normalize + write (heads concatenated: channel = h*64 + dh)
#pragma unroll
  for (int r = 0; r < 8; ++r) {
    const float inv = 1.0f / rsum[r];
    const int qg = qrow_base + r;
    const size_t rowoff = ((size_t)b * S_ + qg) * D_ + h * DH_;
#pragma unroll
    for (int dt = 0; dt < 4; ++dt)
      attn[rowoff + dt * 16 + lr] = (__bf16)(acc[dt][r] * inv);
  }
}

// ---------------------------------------------------------------------------
// Out-proj GEMM: y1 = attn[4096,1024] * w2b[1024,1024]^T + bias + x (residual)
// ---------------------------------------------------------------------------
__global__ __launch_bounds__(256) void out_gemm(const __bf16* __restrict__ ab,
                                                const __bf16* __restrict__ w2b,
                                                const float* __restrict__ bias,
                                                const float* __restrict__ xres,
                                                float* __restrict__ y1) {
  const int m0 = (blockIdx.x / 8) * 128;
  const int n0 = (blockIdx.x % 8) * 128;
  __shared__ __align__(16) __bf16 Asm[2][128][40];
  __shared__ __align__(16) __bf16 Bsm[2][128][40];
  const int t = threadIdx.x;
  const int w = t >> 5, L = t & 31;
  const int wm = w >> 1, wn = w & 1;
  const int half = L >> 4, lr = L & 15;

  v8f acc[2][4] = {};
  gemm_stage(ab, w2b, Asm[0], Bsm[0], m0, n0, 0, t);
  for (int k0 = 0; k0 < 1024; k0 += 32) {
    const int buf = (k0 >> 5) & 1;
    if (k0) __syncthreads();
    if (k0 + 32 < 1024) {
      gemm_stage(ab, w2b, Asm[buf ^ 1], Bsm[buf ^ 1], m0, n0, k0 + 32, t);
      async_wait_le4();
    } else {
      async_wait_0();
    }
    __syncthreads();
    v16bf afr[2], bfr[4];
#pragma unroll
    for (int i = 0; i < 2; ++i) afr[i] = make_afrag(&Asm[buf][wm * 32 + i * 16 + lr][0], half);
#pragma unroll
    for (int j = 0; j < 4; ++j) bfr[j] = make_bfrag(&Bsm[buf][wn * 64 + j * 16 + lr][0], half);
#pragma unroll
    for (int i = 0; i < 2; ++i)
#pragma unroll
      for (int j = 0; j < 4; ++j) acc[i][j] = wmma_bf16(afr[i], bfr[j], acc[i][j]);
  }
#pragma unroll
  for (int j = 0; j < 4; ++j) {
    const int C = n0 + wn * 64 + j * 16 + lr;
    const float bs = bias[C];
#pragma unroll
    for (int i = 0; i < 2; ++i)
#pragma unroll
      for (int r = 0; r < 8; ++r) {
        const int R = m0 + wm * 32 + i * 16 + half * 8 + r;
        const size_t idx = (size_t)R * 1024 + C;
        y1[idx] = acc[i][j][r] + bs + xres[idx];
      }
  }
}

// ---------------------------------------------------------------------------
// LayerNorm over D=1024 per row; optionally adds a second tensor first.
// In-place safe (row cached in registers before any write).
// ---------------------------------------------------------------------------
__global__ __launch_bounds__(256) void ln_kernel(const float* __restrict__ A,
                                                 const float* __restrict__ B2,
                                                 const float* __restrict__ g,
                                                 const float* __restrict__ bb,
                                                 float* __restrict__ out, int addB) {
  __shared__ float red[8];
  const int row = blockIdx.x, t = threadIdx.x;
  const float* a  = A  + (size_t)row * 1024;
  const float* b2 = B2 + (size_t)row * 1024;
  float v[4];
#pragma unroll
  for (int i = 0; i < 4; ++i) {
    const int c = t * 4 + i;
    float x = a[c];
    if (addB) x += b2[c];
    v[i] = x;
  }
  float s = v[0] + v[1] + v[2] + v[3];
#pragma unroll
  for (int m = 1; m < 32; m <<= 1) s += __shfl_xor(s, m, 32);
  if ((t & 31) == 0) red[t >> 5] = s;
  __syncthreads();
  float tot = red[0];
#pragma unroll
  for (int i = 1; i < 8; ++i) tot += red[i];
  const float mu = tot * (1.0f / 1024.0f);
  float q = 0.0f;
#pragma unroll
  for (int i = 0; i < 4; ++i) { const float d = v[i] - mu; q += d * d; }
  __syncthreads();
#pragma unroll
  for (int m = 1; m < 32; m <<= 1) q += __shfl_xor(q, m, 32);
  if ((t & 31) == 0) red[t >> 5] = q;
  __syncthreads();
  float tv = red[0];
#pragma unroll
  for (int i = 1; i < 8; ++i) tv += red[i];
  const float inv = rsqrtf(tv * (1.0f / 1024.0f) + 1e-5f);
#pragma unroll
  for (int i = 0; i < 4; ++i) {
    const int c = t * 4 + i;
    out[(size_t)row * 1024 + c] = (v[i] - mu) * inv * g[c] + bb[c];
  }
}

// ---------------------------------------------------------------------------
// Spectral block: per (b,d) column, 2048-pt FFT in LDS.
// Forward DIF (natural->bitrev), Hermitian filter multiply at bitrev index
// (folds 1/N), inverse DIT (bitrev->natural).  Real part == irfft(rfft*W).
// ---------------------------------------------------------------------------
__global__ __launch_bounds__(256) void fft_spectral(const float* __restrict__ x,
                                                    const float* __restrict__ wr,
                                                    const float* __restrict__ wi,
                                                    float* __restrict__ xwave) {
  __shared__ float re[2048];
  __shared__ float im[2048];
  const int col = blockIdx.x;
  const int b = col >> 10;
  const int d = col & 1023;
  const int t = threadIdx.x;

#pragma unroll
  for (int i = 0; i < 8; ++i) {
    const int s = t + 256 * i;
    re[s] = x[((size_t)b * S_ + s) * D_ + d];
    im[s] = 0.0f;
  }
  __syncthreads();

  // forward DIF: len 2048 -> 2
  for (int sh = 10; sh >= 0; --sh) {
    const int halfl = 1 << sh;
    const int len = halfl << 1;
    const float astep = -6.283185307179586f / (float)len;
#pragma unroll
    for (int i = 0; i < 4; ++i) {
      const int j = t + 256 * i;
      const int grp = j >> sh;
      const int pos = j & (halfl - 1);
      const int i0 = grp * len + pos;
      const int i1 = i0 + halfl;
      float c, sn;
      __sincosf(astep * (float)pos, &sn, &c);
      const float ar = re[i0], ai = im[i0];
      const float br = re[i1], bi = im[i1];
      const float dr = ar - br, di = ai - bi;
      re[i0] = ar + br; im[i0] = ai + bi;
      re[i1] = dr * c - di * sn;
      im[i1] = dr * sn + di * c;
    }
    __syncthreads();
  }

  // filter multiply at bit-reversed positions (Hermitian extension + 1/N)
#pragma unroll
  for (int i = 0; i < 8; ++i) {
    const int p = t + 256 * i;
    const int k = (int)(__brev((unsigned)p) >> (32 - 11));
    float fr, fi;
    if (k <= 1024) { fr = wr[(size_t)k * D_ + d]; fi = wi[(size_t)k * D_ + d]; }
    else { const int km = 2048 - k; fr = wr[(size_t)km * D_ + d]; fi = -wi[(size_t)km * D_ + d]; }
    const float scale = 1.0f / 2048.0f;
    const float a = re[p], bb = im[p];
    re[p] = (a * fr - bb * fi) * scale;
    im[p] = (a * fi + bb * fr) * scale;
  }
  __syncthreads();

  // inverse DIT: len 2 -> 2048, conjugate twiddles
  for (int sh = 0; sh <= 10; ++sh) {
    const int halfl = 1 << sh;
    const int len = halfl << 1;
    const float astep = 6.283185307179586f / (float)len;
#pragma unroll
    for (int i = 0; i < 4; ++i) {
      const int j = t + 256 * i;
      const int grp = j >> sh;
      const int pos = j & (halfl - 1);
      const int i0 = grp * len + pos;
      const int i1 = i0 + halfl;
      float c, sn;
      __sincosf(astep * (float)pos, &sn, &c);
      const float br = re[i1], bi = im[i1];
      const float tr = br * c - bi * sn;
      const float ti = br * sn + bi * c;
      const float ar = re[i0], ai = im[i0];
      re[i1] = ar - tr; im[i1] = ai - ti;
      re[i0] = ar + tr; im[i0] = ai + ti;
    }
    __syncthreads();
  }

#pragma unroll
  for (int i = 0; i < 8; ++i) {
    const int s = t + 256 * i;
    xwave[((size_t)b * S_ + s) * D_ + d] = re[s];
  }
}

// ---------------------------------------------------------------------------
// Launch
// ---------------------------------------------------------------------------
extern "C" void kernel_launch(void* const* d_in, const int* in_sizes, int n_in,
                              void* d_out, int out_size, void* d_ws, size_t ws_size,
                              hipStream_t stream) {
  const float* x   = (const float*)d_in[0];
  const float* w1  = (const float*)d_in[1];
  const float* b1  = (const float*)d_in[2];
  const float* w2  = (const float*)d_in[3];
  const float* b2  = (const float*)d_in[4];
  const float* g1  = (const float*)d_in[5];
  const float* bb1 = (const float*)d_in[6];
  const float* swr = (const float*)d_in[7];
  const float* swi = (const float*)d_in[8];
  const float* g2  = (const float*)d_in[9];
  const float* bb2 = (const float*)d_in[10];
  float* out = (float*)d_out;

  char* ws = (char*)d_ws;
  size_t off = 0;
  auto carve = [&](size_t bytes) -> char* {
    char* p = ws + off;
    off += (bytes + 255) & ~(size_t)255;
    return p;
  };
  __bf16* xb    = (__bf16*)carve((size_t)NTOK * D_ * 2);      // also reused as attn out
  __bf16* w1b   = (__bf16*)carve((size_t)3 * D_ * D_ * 2);
  __bf16* w2b   = (__bf16*)carve((size_t)D_ * D_ * 2);
  __bf16* Qb    = (__bf16*)carve((size_t)NTOK * D_ * 2);
  __bf16* Kb    = (__bf16*)carve((size_t)NTOK * D_ * 2);
  __bf16* Vb    = (__bf16*)carve((size_t)NTOK * D_ * 2);
  float*  y1    = (float*)carve((size_t)NTOK * D_ * 4);
  float*  xwave = (float*)carve((size_t)NTOK * D_ * 4);
  (void)ws_size; (void)in_sizes; (void)n_in; (void)out_size;

  // 1) bf16 copies of activations and weights
  cvt_bf16<<<4096, 256, 0, stream>>>(x,  xb,  NTOK * D_);
  cvt_bf16<<<3072, 256, 0, stream>>>(w1, w1b, 3 * D_ * D_);
  cvt_bf16<<<1024, 256, 0, stream>>>(w2, w2b, D_ * D_);

  // 2) QKV projection (WMMA + async LDS pipeline) -> Q/K/V [B,H,S,DH] bf16
  qkv_gemm<<<32 * 24, 256, 0, stream>>>(xb, w1b, b1, Qb, Kb, Vb);

  // 3) banded flash attention (WMMA) -> attn bf16 (reuses xb region)
  attn_kernel<<<2 * 16 * 16, 256, 0, stream>>>(Qb, Kb, Vb, xb);

  // 4) out-proj (WMMA + async LDS pipeline) + bias + residual -> y1 f32
  out_gemm<<<32 * 8, 256, 0, stream>>>(xb, w2b, b2, x, y1);

  // 5) LN1 in-place
  ln_kernel<<<NTOK, 256, 0, stream>>>(y1, y1, g1, bb1, y1, 0);

  // 6) spectral filter via LDS FFT -> xwave
  fft_spectral<<<2 * D_, 256, 0, stream>>>(x, swr, swi, xwave);

  // 7) final mix + LN2 -> out
  ln_kernel<<<NTOK, 256, 0, stream>>>(y1, xwave, g2, bb2, out, 1);
}
